// FlashGemma2Attention_54520314856263
// MI455X (gfx1250) — matrix operand order repaired and checked by reading, not compiled, not run on hardware
//
#include <hip/hip_runtime.h>

// ---------------------------------------------------------------- types
typedef __bf16 bf16;
typedef __attribute__((ext_vector_type(16))) bf16 v16bf;
typedef __attribute__((ext_vector_type(8)))  bf16 v8bf;
typedef __attribute__((ext_vector_type(8)))  float v8f;

union Frag { v16bf v; v8bf h[2]; };

__device__ __forceinline__ v8f v8f_zero() { v8f z = {0.f,0.f,0.f,0.f,0.f,0.f,0.f,0.f}; return z; }

__device__ __forceinline__ v8f wmma_bf16(const Frag& a, const Frag& b, v8f c) {
    return __builtin_amdgcn_wmma_f32_16x16x32_bf16(false, a.v, false, b.v,
                                                   (short)0, c, false, false);
}

// CDNA5 async global->LDS copy (ASYNCcnt-tracked, no VGPR staging).
// Generic pointers to __shared__ carry the LDS byte offset in their low 32 bits.
__device__ __forceinline__ void async_load_b128(const void* gsrc, void* ldst) {
    unsigned     lds = (unsigned)(unsigned long long)ldst;
    unsigned long long ga = (unsigned long long)gsrc;
    asm volatile("global_load_async_to_lds_b128 %0, %1, off"
                 :: "v"(lds), "v"(ga) : "memory");
}
__device__ __forceinline__ void wait_async() {
    asm volatile("s_wait_asynccnt 0" ::: "memory");
}

// ---------------------------------------------------------------- sizes
#define T_SZ   4096
#define S_SZ   2048
#define B_SZ   2
#define HID_SZ 3584
#define NH     16
#define NKV    8
#define HD     256
#define QKV_N  8192          // (NH + 2*NKV) * HD
#define WIN    1024
#define SCALE_F 0.0625f      // 256^-0.5
#define CAP_F   50.0f

// ---------------------------------------------------------------- fp32 -> bf16
__global__ void f32_to_bf16_kernel(const float* __restrict__ src,
                                   bf16* __restrict__ dst, size_t n)
{
    size_t i = (size_t)blockIdx.x * blockDim.x + threadIdx.x;
    size_t stride = (size_t)gridDim.x * blockDim.x;
    for (; i < n; i += stride) dst[i] = (bf16)src[i];
}

// ---------------------------------------------------------------- GEMM  D = A (MxK) * W(NxK)^T
// A, W bf16 K-contiguous.  Double-buffered LDS, async global->LDS staging.
// If outF != null: write fp32 MxN.  Else: split columns into q/k/v bf16 buffers.
__global__ void __launch_bounds__(256)
gemm_bf16_nt(const bf16* __restrict__ A, const bf16* __restrict__ W,
             int M, int N, int K,
             float* __restrict__ outF,
             bf16* __restrict__ outq, bf16* __restrict__ outk, bf16* __restrict__ outv)
{
    __shared__ alignas(16) bf16 As[2][128][40];   // 32 K-cols + 8 pad
    __shared__ alignas(16) bf16 Bs[2][128][40];

    const int tid  = threadIdx.x;
    const int lane = tid & 31;
    const int lr   = lane & 15;      // 0..15
    const int hs   = lane >> 4;      // 0/1 lane-half
    const int wave = tid >> 5;       // 0..7
    const int wm   = wave >> 2;      // 0..1 (64 rows each)
    const int wn   = wave & 3;       // 0..3 (32 cols each)
    const int m0   = blockIdx.y * 128;
    const int n0   = blockIdx.x * 128;

    // staging coords: 512 16B-chunks per tile, 2 per thread (for A and for B each)
    const int r0 = tid >> 2,         c0 = (tid & 3) << 3;
    const int r1 = (tid + 256) >> 2, c1 = ((tid + 256) & 3) << 3;

    v8f acc[4][2];
#pragma unroll
    for (int mt = 0; mt < 4; ++mt)
#pragma unroll
        for (int nt = 0; nt < 2; ++nt) acc[mt][nt] = v8f_zero();

    // prologue: stage buffer 0
    async_load_b128(&A[(size_t)(m0 + r0) * K + c0], &As[0][r0][c0]);
    async_load_b128(&W[(size_t)(n0 + r0) * K + c0], &Bs[0][r0][c0]);
    async_load_b128(&A[(size_t)(m0 + r1) * K + c1], &As[0][r1][c1]);
    async_load_b128(&W[(size_t)(n0 + r1) * K + c1], &Bs[0][r1][c1]);

    int cur = 0;
    for (int k0 = 0; k0 < K; k0 += 32) {
        wait_async();
        __syncthreads();                       // current buffer visible to all waves
        if (k0 + 32 < K) {                     // prefetch next stage into other buffer
            const int kn = k0 + 32;
            const int nb = cur ^ 1;
            async_load_b128(&A[(size_t)(m0 + r0) * K + kn + c0], &As[nb][r0][c0]);
            async_load_b128(&W[(size_t)(n0 + r0) * K + kn + c0], &Bs[nb][r0][c0]);
            async_load_b128(&A[(size_t)(m0 + r1) * K + kn + c1], &As[nb][r1][c1]);
            async_load_b128(&W[(size_t)(n0 + r1) * K + kn + c1], &Bs[nb][r1][c1]);
        }
        Frag a[4], b[2];
#pragma unroll
        for (int mt = 0; mt < 4; ++mt) {       // A frag: lane row = lr, K 0-7/16-23 per half
            const bf16* p = &As[cur][wm*64 + mt*16 + lr][hs*8];
            a[mt].h[0] = *(const v8bf*)p;
            a[mt].h[1] = *(const v8bf*)(p + 16);
        }
#pragma unroll
        for (int nt = 0; nt < 2; ++nt) {       // B frag: lane col = lr, K 0-7/16-23 per half
            const bf16* p = &Bs[cur][wn*32 + nt*16 + lr][hs*8];
            b[nt].h[0] = *(const v8bf*)p;
            b[nt].h[1] = *(const v8bf*)(p + 16);
        }
#pragma unroll
        for (int mt = 0; mt < 4; ++mt)
#pragma unroll
            for (int nt = 0; nt < 2; ++nt)
                acc[mt][nt] = wmma_bf16(a[mt], b[nt], acc[mt][nt]);
        cur ^= 1;
    }

    // epilogue: C layout -> element r: m = r + 8*hs (lane half), n = lr
#pragma unroll
    for (int mt = 0; mt < 4; ++mt) {
#pragma unroll
        for (int nt = 0; nt < 2; ++nt) {
#pragma unroll
            for (int r = 0; r < 8; ++r) {
                int m = m0 + wm*64 + mt*16 + r + hs*8;
                int n = n0 + wn*32 + nt*16 + lr;
                float val = acc[mt][nt][r];
                if (outF) {
                    outF[(size_t)m * N + n] = val;
                } else {
                    bf16 bv = (bf16)val;
                    if (n < NH*HD)            outq[(size_t)m*(NH*HD)  + n] = bv;
                    else if (n < (NH+NKV)*HD) outk[(size_t)m*(NKV*HD) + (n - NH*HD)] = bv;
                    else                      outv[(size_t)m*(NKV*HD) + (n - (NH+NKV)*HD)] = bv;
                }
            }
        }
    }
}

// ---------------------------------------------------------------- RoPE (in-place on bf16 q,k)
__global__ void rope_kernel(bf16* __restrict__ q, bf16* __restrict__ k,
                            const float* __restrict__ cs, const float* __restrict__ sn)
{
    const size_t nq = (size_t)T_SZ * NH  * (HD/2);
    const size_t nk = (size_t)T_SZ * NKV * (HD/2);
    const size_t total = nq + nk;
    size_t idx = (size_t)blockIdx.x * blockDim.x + threadIdx.x;
    size_t stride = (size_t)gridDim.x * blockDim.x;
    for (; idx < total; idx += stride) {
        bf16* buf; int heads; size_t id = idx;
        if (id < nq) { buf = q; heads = NH; }
        else         { id -= nq; buf = k; heads = NKV; }
        int i  = (int)(id % (HD/2)); id /= (HD/2);
        int hh = (int)(id % heads);
        size_t t = id / heads;
        size_t base = (t * (size_t)heads + hh) * HD;
        float c = cs[t * (HD/2) + i];
        float s = sn[t * (HD/2) + i];
        float x1 = (float)buf[base + i];
        float x2 = (float)buf[base + i + HD/2];
        buf[base + i]        = (bf16)(x1 * c - x2 * s);
        buf[base + i + HD/2] = (bf16)(x2 * c + x1 * s);
    }
}

// ---------------------------------------------------------------- V transpose: [t][kv][d] -> [b*kv][d][s]
__global__ void vtrans_kernel(const bf16* __restrict__ v, bf16* __restrict__ vt)
{
    const size_t total = (size_t)T_SZ * NKV * HD;
    size_t idx = (size_t)blockIdx.x * blockDim.x + threadIdx.x;
    size_t stride = (size_t)gridDim.x * blockDim.x;
    for (; idx < total; idx += stride) {
        int d = (int)(idx % HD);
        size_t rest = idx / HD;
        int kvh = (int)(rest % NKV);
        size_t t = rest / NKV;
        int bb = (int)(t / S_SZ);
        int ss = (int)(t % S_SZ);
        vt[((size_t)(bb*NKV + kvh) * HD + d) * S_SZ + ss] = v[idx];
    }
}

// ---------------------------------------------------------------- flash attention
// grid (S/64, H, B), 128 threads = 4 waves, each wave owns 16 query rows.
__global__ void __launch_bounds__(128)
attn_kernel(const bf16* __restrict__ Q,    // [T][H][D]
            const bf16* __restrict__ Kc,   // [T][KV][D]
            const bf16* __restrict__ Vt,   // [B*KV][D][S]
            bf16* __restrict__ Oout)       // [T][H*D]
{
    __shared__ alignas(16) bf16 Ks[32][264];    // 32 keys x 256 d  (+8 pad)
    __shared__ alignas(16) bf16 Vts[256][40];   // 256 d  x 32 keys (+8 pad)
    __shared__ alignas(16) bf16 Ps[4][16][40];  // per-wave P transpose scratch

    const int qt  = blockIdx.x;
    const int h   = blockIdx.y;
    const int b   = blockIdx.z;
    const int kvh = h >> 1;                 // H/KV = 2
    const int qs  = qt * 64;

    const int tid  = threadIdx.x;
    const int lane = tid & 31;
    const int lr   = lane & 15;
    const int hs   = lane >> 4;
    const int wave = tid >> 5;              // 0..3

    // Q A-fragments, register resident: row = lr within wave tile
    Frag qf[8];
    {
        const bf16* qbase = Q + ((size_t)(b*S_SZ + qs + wave*16 + lr) * (NH*HD) + h*HD);
#pragma unroll
        for (int kc = 0; kc < 8; ++kc) {
            const bf16* p = qbase + kc*32 + hs*8;
            qf[kc].h[0] = *(const v8bf*)p;
            qf[kc].h[1] = *(const v8bf*)(p + 16);
        }
    }

    v8f o[16];
#pragma unroll
    for (int dt = 0; dt < 16; ++dt) o[dt] = v8f_zero();
    float rm[8], rs[8];
#pragma unroll
    for (int r = 0; r < 8; ++r) { rm[r] = -1e30f; rs[r] = 0.f; }

    int ks_lo = qs - WIN; if (ks_lo < 0) ks_lo = 0;

    for (int ks = ks_lo; ks < qs + 64; ks += 32) {
        __syncthreads();                    // all waves done reading previous tiles
#pragma unroll
        for (int i = 0; i < 8; ++i) {       // K tile: 32 rows x 32 chunks, async to LDS
            int idx = tid + 128*i;
            int row = idx >> 5;
            int col = (idx & 31) << 3;
            async_load_b128(&Kc[(size_t)(b*S_SZ + ks + row) * (NKV*HD) + kvh*HD + col],
                            &Ks[row][col]);
        }
#pragma unroll
        for (int i = 0; i < 8; ++i) {       // V^T tile: 256 rows x 4 chunks, async to LDS
            int idx = tid + 128*i;
            int row = idx >> 2;
            int col = (idx & 3) << 3;
            async_load_b128(&Vt[((size_t)(b*NKV + kvh) * HD + row) * S_SZ + ks + col],
                            &Vts[row][col]);
        }
        wait_async();
        __syncthreads();                    // tiles visible to all waves

        // ---- scores S = Q K^T : 2 key-tiles x 8 d-chunks
        v8f sc[2];
#pragma unroll
        for (int nt = 0; nt < 2; ++nt) {
            sc[nt] = v8f_zero();
#pragma unroll
            for (int kc = 0; kc < 8; ++kc) {
                Frag kf;
                const bf16* p = &Ks[nt*16 + lr][kc*32 + hs*8];
                kf.h[0] = *(const v8bf*)p;
                kf.h[1] = *(const v8bf*)(p + 16);
                sc[nt] = wmma_bf16(qf[kc], kf, sc[nt]);
            }
        }

        // ---- softcap + window mask + online softmax (row m lives in one 16-lane half)
        float p2[2][8];
#pragma unroll
        for (int r = 0; r < 8; ++r) {
            const int iq = qs + wave*16 + r + hs*8;
            float mx = -1e30f;
#pragma unroll
            for (int nt = 0; nt < 2; ++nt) {
                int j = ks + nt*16 + lr;
                float x = sc[nt][r] * (SCALE_F / CAP_F);
                x = fminf(fmaxf(x, -20.f), 20.f);
                float e2 = __expf(2.f * x);
                float s = CAP_F * ((e2 - 1.f) / (e2 + 1.f));   // softcap * tanh
                bool ok = (j <= iq) && (j + WIN >= iq);
                s = ok ? s : -1e30f;
                p2[nt][r] = s;
                mx = fmaxf(mx, s);
            }
#pragma unroll
            for (int d = 1; d < 16; d <<= 1)
                mx = fmaxf(mx, __shfl_xor(mx, d, 16));
            float mnew  = fmaxf(rm[r], mx);
            float alpha = __expf(rm[r] - mnew);
            rm[r] = mnew;
            float lsum = 0.f;
#pragma unroll
            for (int nt = 0; nt < 2; ++nt) {
                float e = (p2[nt][r] > -1e29f) ? __expf(p2[nt][r] - mnew) : 0.f;
                p2[nt][r] = e;
                lsum += e;
            }
#pragma unroll
            for (int d = 1; d < 16; d <<= 1)
                lsum += __shfl_xor(lsum, d, 16);
            rs[r] = rs[r] * alpha + lsum;
#pragma unroll
            for (int dt = 0; dt < 16; ++dt)
                o[dt][r] *= alpha;
        }

        // ---- P (C layout) -> per-wave LDS -> A fragment (wave-internal, DS is in-order)
#pragma unroll
        for (int r = 0; r < 8; ++r) {
            int m = r + hs*8;
            Ps[wave][m][lr]      = (bf16)p2[0][r];
            Ps[wave][m][16 + lr] = (bf16)p2[1][r];
        }
        Frag pf;
        {
            const bf16* p = &Ps[wave][lr][hs*8];
            pf.h[0] = *(const v8bf*)p;
            pf.h[1] = *(const v8bf*)(p + 16);
        }

        // ---- O += P V : 16 d-tiles, contract 32 keys
#pragma unroll
        for (int dt = 0; dt < 16; ++dt) {
            Frag vf;
            const bf16* p = &Vts[dt*16 + lr][hs*8];
            vf.h[0] = *(const v8bf*)p;
            vf.h[1] = *(const v8bf*)(p + 16);
            o[dt] = wmma_bf16(pf, vf, o[dt]);
        }
    }

    // ---- normalize + store bf16 [t][h*256+d]
#pragma unroll
    for (int r = 0; r < 8; ++r) {
        float inv = 1.f / rs[r];
        size_t rowoff = (size_t)(b*S_SZ + qs + wave*16 + r + hs*8) * (NH*HD) + h*HD;
#pragma unroll
        for (int dt = 0; dt < 16; ++dt)
            Oout[rowoff + dt*16 + lr] = (bf16)(o[dt][r] * inv);
    }
}

// ---------------------------------------------------------------- launcher
extern "C" void kernel_launch(void* const* d_in, const int* in_sizes, int n_in,
                              void* d_out, int out_size, void* d_ws, size_t ws_size,
                              hipStream_t stream)
{
    (void)in_sizes; (void)n_in; (void)out_size; (void)ws_size;
    const float* hidden = (const float*)d_in[0];
    const float* cosb   = (const float*)d_in[1];
    const float* sinb   = (const float*)d_in[2];
    const float* wqkv   = (const float*)d_in[3];
    const float* wo     = (const float*)d_in[4];
    float* out = (float*)d_out;

    char* base = (char*)d_ws;
    size_t off = 0;
    auto carve = [&](size_t elems) -> bf16* {
        bf16* p = (bf16*)(base + off);
        off += ((elems * sizeof(bf16)) + 255) & ~(size_t)255;
        return p;
    };
    bf16* hiddenBF = carve((size_t)T_SZ * HID_SZ);        // 29.4 MB
    bf16* wqkvBF   = carve((size_t)QKV_N * HID_SZ);       // 58.7 MB
    bf16* woBF     = carve((size_t)HID_SZ * NH * HD);     // 29.4 MB
    bf16* qBF      = carve((size_t)T_SZ * NH  * HD);      // 33.6 MB
    bf16* kBF      = carve((size_t)T_SZ * NKV * HD);      // 16.8 MB
    bf16* vBF      = carve((size_t)T_SZ * NKV * HD);      // 16.8 MB
    bf16* vTBF     = carve((size_t)T_SZ * NKV * HD);      // 16.8 MB
    bf16* attnBF   = carve((size_t)T_SZ * NH  * HD);      // 33.6 MB

    f32_to_bf16_kernel<<<2048, 256, 0, stream>>>(hidden, hiddenBF, (size_t)T_SZ * HID_SZ);
    f32_to_bf16_kernel<<<2048, 256, 0, stream>>>(wqkv,   wqkvBF,   (size_t)QKV_N * HID_SZ);
    f32_to_bf16_kernel<<<2048, 256, 0, stream>>>(wo,     woBF,     (size_t)HID_SZ * NH * HD);

    {   // QKV projection: (T x HID) * (8192 x HID)^T -> q/k/v bf16
        dim3 g(QKV_N / 128, T_SZ / 128);
        gemm_bf16_nt<<<g, 256, 0, stream>>>(hiddenBF, wqkvBF, T_SZ, QKV_N, HID_SZ,
                                            nullptr, qBF, kBF, vBF);
    }

    rope_kernel<<<4096, 256, 0, stream>>>(qBF, kBF, cosb, sinb);
    vtrans_kernel<<<4096, 256, 0, stream>>>(vBF, vTBF);

    {   // flash attention
        dim3 g(S_SZ / 64, NH, B_SZ);
        attn_kernel<<<g, 128, 0, stream>>>(qBF, kBF, vTBF, attnBF);
    }

    {   // output projection: (T x 4096) * (3584 x 4096)^T -> fp32 out
        dim3 g(HID_SZ / 128, T_SZ / 128);
        gemm_bf16_nt<<<g, 256, 0, stream>>>(attnBF, woBF, T_SZ, HID_SZ, NH * HD,
                                            out, nullptr, nullptr, nullptr);
    }
}
